// LightweightConvolution_1185410974336
// MI455X (gfx1250) — matrix operand order repaired and checked by reading, not compiled, not run on hardware
//
#include <hip/hip_runtime.h>
#include <hip/hip_bf16.h>
#include <math.h>

typedef __bf16 v16bf __attribute__((ext_vector_type(16)));
typedef float  v8f   __attribute__((ext_vector_type(8)));

#define BDIM 256
#define BM 128
#define BN 64
#define BK 32
#define KP 16          // bf16 pairs (dwords) per K-step
#define LDA 20         // padded LDS row stride in dwords (16B aligned, conflict-free)
#define BUFB 20480     // LDS double-buffer stride in bytes (2*128*20*4 == 2*2*64*20*4)

union FragBF { unsigned u[8]; uint4 q[2]; v16bf v; };

__device__ __forceinline__ unsigned f2bf_rne(float f) {
    unsigned u = __float_as_uint(f);
    unsigned r = u + 0x7FFFu + ((u >> 16) & 1u);
    return r >> 16;
}

__device__ __forceinline__ void split_hi_lo(float x, unsigned &hi, unsigned &lo) {
    unsigned h = f2bf_rne(x);
    float hf = __uint_as_float(h << 16);
    hi = h;
    lo = f2bf_rne(x - hf);
}

__device__ __forceinline__ void pack_pair(float x0, float x1, unsigned &phi, unsigned &plo) {
    unsigned h0, l0, h1, l1;
    split_hi_lo(x0, h0, l0);
    split_hi_lo(x1, h1, l1);
    phi = h0 | (h1 << 16);
    plo = l0 | (l1 << 16);
}

__device__ __forceinline__ v8f vzero8() {
    v8f z = {0.f, 0.f, 0.f, 0.f, 0.f, 0.f, 0.f, 0.f};
    return z;
}

// async global -> LDS copy of 16 bytes per lane (gfx1250, ASYNCcnt-tracked)
__device__ __forceinline__ void async_b128(unsigned lds_addr, const unsigned* gaddr) {
    asm volatile("global_load_async_to_lds_b128 %0, %1, off"
                 :: "v"(lds_addr), "v"(gaddr) : "memory");
}
__device__ __forceinline__ void wait_async0() {
    asm volatile("s_wait_asynccnt 0x0" ::: "memory");
}
__device__ __forceinline__ unsigned lds_addr_of(const void* p) {
    return (unsigned)(unsigned long long)p;   // addr[31:0] of generic LDS ptr = LDS offset
}

// ---------------------------------------------------------------------------
// Kernel 0: one-shot fp32 -> packed {bf16 hi, bf16 lo} pair split
// ---------------------------------------------------------------------------
__global__ __launch_bounds__(256)
void lc_split(const float* __restrict__ x, unsigned* __restrict__ hi,
              unsigned* __restrict__ lo, int npairs) {
    int i = blockIdx.x * 256 + threadIdx.x;
    if (i < npairs) {
        float2 v = ((const float2*)x)[i];
        unsigned ph, pl;
        pack_pair(v.x, v.y, ph, pl);
        hi[i] = ph;
        lo[i] = pl;
    }
}

// ---------------------------------------------------------------------------
// Kernel 1: softmax-normalize conv kernels (H=4, K=31)
// ---------------------------------------------------------------------------
__global__ void lc_softmax_rows(const float* __restrict__ w, float* __restrict__ ws,
                                int Hn, int Kn) {
    int h = threadIdx.x;
    if (h < Hn) {
        float mx = -1e30f;
        for (int k = 0; k < Kn; ++k) mx = fmaxf(mx, w[h * Kn + k]);
        float s = 0.f;
        for (int k = 0; k < Kn; ++k) s += __expf(w[h * Kn + k] - mx);
        float inv = 1.f / s;
        for (int k = 0; k < Kn; ++k) ws[h * Kn + k] = __expf(w[h * Kn + k] - mx) * inv;
    }
}

// ---------------------------------------------------------------------------
// Kernel 2: xg = GLU(q @ W1^T + b1)
// bf16x3 WMMA; async global->LDS double-buffered pipeline (no data registers)
// ---------------------------------------------------------------------------
__global__ __launch_bounds__(BDIM, 1)
void lc_gemm1_glu(const unsigned* __restrict__ AH, const unsigned* __restrict__ AL,
                  const unsigned* __restrict__ BH, const unsigned* __restrict__ BL,
                  const float* __restrict__ b1, float* __restrict__ xg, int Cdim) {
    // [buf][hi/lo][row][pair] and [buf][hh][hi/lo][row][pair]
    __shared__ __align__(16) unsigned As[2][2][BM][LDA];
    __shared__ __align__(16) unsigned Bs[2][2][2][BN][LDA];

    const int tid  = threadIdx.x;
    const int lane = tid & 31;
    const int wave = tid >> 5;
    const int wm   = wave & 3;
    const int wn   = wave >> 2;
    const int r    = lane & 15;
    const int half = lane >> 4;
    const int Cp   = Cdim >> 1;

    const int n0 = blockIdx.x * BN;
    const int m0 = blockIdx.y * BM;

    // per-thread async chunk addressing (4 A-chunks, 4 B-chunks of b128)
    unsigned aLds[4], bLds[4];
    const unsigned* gA[4];
    const unsigned* gB[4];
    #pragma unroll
    for (int it = 0; it < 4; ++it) {
        int idx = tid + it * BDIM;               // 0..1023
        int split = idx >> 9;
        int rem   = idx & 511;
        int row   = rem >> 2;
        int q4    = rem & 3;
        aLds[it] = lds_addr_of(&As[0][split][row][q4 * 4]);
        gA[it]   = (split ? AL : AH) + (size_t)(m0 + row) * Cp + q4 * 4;
    }
    #pragma unroll
    for (int it = 0; it < 4; ++it) {
        int idx = tid + it * BDIM;               // 0..1023
        int hh    = idx >> 9;
        int rem   = idx & 511;
        int split = rem >> 8;
        int rem2  = rem & 255;
        int row   = rem2 >> 2;
        int q4    = rem2 & 3;
        bLds[it] = lds_addr_of(&Bs[0][hh][split][row][q4 * 4]);
        gB[it]   = (split ? BL : BH) + (size_t)(hh * Cdim + n0 + row) * Cp + q4 * 4;
    }

    auto prefetch = [&](int buf) {
        #pragma unroll
        for (int it = 0; it < 4; ++it) async_b128(aLds[it] + buf * BUFB, gA[it]);
        #pragma unroll
        for (int it = 0; it < 4; ++it) async_b128(bLds[it] + buf * BUFB, gB[it]);
        #pragma unroll
        for (int it = 0; it < 4; ++it) { gA[it] += KP; gB[it] += KP; }
    };

    v8f acc[2][2][2];  // [a/g half][mi][ni]
    #pragma unroll
    for (int h = 0; h < 2; ++h)
        #pragma unroll
        for (int mi = 0; mi < 2; ++mi)
            #pragma unroll
            for (int ni = 0; ni < 2; ++ni) acc[h][mi][ni] = vzero8();

    const int NS = Cdim / BK;
    prefetch(0);
    wait_async0();
    __syncthreads();

    for (int s = 0; s < NS; ++s) {
        const int p = s & 1;
        if (s + 1 < NS) prefetch(1 - p);   // async loads land in other buffer during WMMA

        FragBF aH[2], aL[2];
        #pragma unroll
        for (int mi = 0; mi < 2; ++mi) {
            int row = wm * 32 + mi * 16 + r;
            aH[mi].q[0] = *(const uint4*)&As[p][0][row][half * 4];
            aH[mi].q[1] = *(const uint4*)&As[p][0][row][half * 4 + 8];
            aL[mi].q[0] = *(const uint4*)&As[p][1][row][half * 4];
            aL[mi].q[1] = *(const uint4*)&As[p][1][row][half * 4 + 8];
        }
        #pragma unroll
        for (int hh = 0; hh < 2; ++hh) {
            #pragma unroll
            for (int ni = 0; ni < 2; ++ni) {
                FragBF bH, bL;
                int col = wn * 32 + ni * 16 + r;
                bH.q[0] = *(const uint4*)&Bs[p][hh][0][col][half * 4];
                bH.q[1] = *(const uint4*)&Bs[p][hh][0][col][half * 4 + 8];
                bL.q[0] = *(const uint4*)&Bs[p][hh][1][col][half * 4];
                bL.q[1] = *(const uint4*)&Bs[p][hh][1][col][half * 4 + 8];
                #pragma unroll
                for (int mi = 0; mi < 2; ++mi) {
                    v8f c = acc[hh][mi][ni];
                    c = __builtin_amdgcn_wmma_f32_16x16x32_bf16(false, aH[mi].v, false, bH.v,
                                                                (short)0, c, false, false);
                    c = __builtin_amdgcn_wmma_f32_16x16x32_bf16(false, aH[mi].v, false, bL.v,
                                                                (short)0, c, false, false);
                    c = __builtin_amdgcn_wmma_f32_16x16x32_bf16(false, aL[mi].v, false, bH.v,
                                                                (short)0, c, false, false);
                    acc[hh][mi][ni] = c;
                }
            }
        }
        wait_async0();       // my tile s+1 loads complete
        __syncthreads();     // all waves done reading buf p & loading buf 1-p
    }

    // epilogue: bias + GLU  (C/D layout: VGPR v -> M = v + 8*half, N = lane%16)
    #pragma unroll
    for (int mi = 0; mi < 2; ++mi) {
        #pragma unroll
        for (int ni = 0; ni < 2; ++ni) {
            int col = n0 + wn * 32 + ni * 16 + r;
            float ba = b1[col];
            float bg = b1[col + Cdim];
            #pragma unroll
            for (int v = 0; v < 8; ++v) {
                int row = m0 + wm * 32 + mi * 16 + half * 8 + v;
                float a = acc[0][mi][ni][v] + ba;
                float g = acc[1][mi][ni][v] + bg;
                float sg = 1.0f / (1.0f + __expf(-g));
                xg[(size_t)row * Cdim + col] = a * sg;
            }
        }
    }
}

// ---------------------------------------------------------------------------
// Kernel 3: depthwise conv along T (K=31), channel c uses kernel c%4,
// + bias + mask; writes y directly as split {hi,lo} bf16 pairs for GEMM2.
// ---------------------------------------------------------------------------
#define CT 64
#define CC 64
#define KHALF 15
__global__ __launch_bounds__(256)
void lc_dwconv(const float* __restrict__ xg, const float* __restrict__ wsoft,
               const float* __restrict__ bias, const int* __restrict__ mask,
               unsigned* __restrict__ yH, unsigned* __restrict__ yL,
               int T, int Cdim) {
    __shared__ float tile[CT + 2 * KHALF][CC];
    __shared__ float wl[4][31];

    const int b   = blockIdx.z;
    const int c0  = blockIdx.y * CC;
    const int t0  = blockIdx.x * CT;
    const int tid = threadIdx.x;
    const int Cp  = Cdim >> 1;

    if (tid < 124) wl[tid / 31][tid % 31] = wsoft[tid];

    for (int idx = tid; idx < (CT + 2 * KHALF) * CC; idx += 256) {
        int rr = idx / CC, cl = idx % CC;
        int t = t0 - KHALF + rr;
        float v = 0.f;
        if (t >= 0 && t < T) v = xg[((size_t)b * T + t) * Cdim + c0 + cl];
        tile[rr][cl] = v;
    }
    __syncthreads();

    const int pl = tid & 31;          // channel pair within tile
    const int tg = tid >> 5;          // 8 t-groups
    const int c  = c0 + 2 * pl;
    const int h0 = (2 * pl) & 3;
    const int h1 = h0 + 1;
    const float bv0 = bias[c];
    const float bv1 = bias[c + 1];

    #pragma unroll 1
    for (int i = 0; i < CT / 8; ++i) {
        int tl = tg + i * 8;
        int t  = t0 + tl;
        float s0 = 0.f, s1 = 0.f;
        #pragma unroll
        for (int k = 0; k < 31; ++k) {
            float2 v = *(const float2*)&tile[tl + k][2 * pl];
            s0 += wl[h0][k] * v.x;
            s1 += wl[h1][k] * v.y;
        }
        int mv = mask[(size_t)b * T + t];
        unsigned ph = 0u, plo = 0u;
        if (mv != 0) pack_pair(s0 + bv0, s1 + bv1, ph, plo);
        size_t o = ((size_t)b * T + t) * Cp + (c0 >> 1) + pl;
        yH[o] = ph;
        yL[o] = plo;
    }
}

// ---------------------------------------------------------------------------
// Kernel 4: out = y @ W2^T + b2 -- bf16x3 WMMA, async double-buffered pipeline
// ---------------------------------------------------------------------------
__global__ __launch_bounds__(BDIM, 1)
void lc_gemm2(const unsigned* __restrict__ AH, const unsigned* __restrict__ AL,
              const unsigned* __restrict__ BH, const unsigned* __restrict__ BL,
              const float* __restrict__ b2, float* __restrict__ out, int Cdim) {
    __shared__ __align__(16) unsigned As[2][2][BM][LDA];
    __shared__ __align__(16) unsigned Bs[2][2][BN][LDA];

    const int tid  = threadIdx.x;
    const int lane = tid & 31;
    const int wave = tid >> 5;
    const int wm   = wave & 3;
    const int wn   = wave >> 2;
    const int r    = lane & 15;
    const int half = lane >> 4;
    const int Cp   = Cdim >> 1;

    const int n0 = blockIdx.x * BN;
    const int m0 = blockIdx.y * BM;

    unsigned aLds[4], bLds[2];
    const unsigned* gA[4];
    const unsigned* gB[2];
    #pragma unroll
    for (int it = 0; it < 4; ++it) {
        int idx = tid + it * BDIM;
        int split = idx >> 9;
        int rem   = idx & 511;
        int row   = rem >> 2;
        int q4    = rem & 3;
        aLds[it] = lds_addr_of(&As[0][split][row][q4 * 4]);
        gA[it]   = (split ? AL : AH) + (size_t)(m0 + row) * Cp + q4 * 4;
    }
    #pragma unroll
    for (int it = 0; it < 2; ++it) {
        int idx = tid + it * BDIM;               // 0..511
        int split = idx >> 8;
        int rem   = idx & 255;
        int row   = rem >> 2;
        int q4    = rem & 3;
        bLds[it] = lds_addr_of(&Bs[0][split][row][q4 * 4]);
        gB[it]   = (split ? BL : BH) + (size_t)(n0 + row) * Cp + q4 * 4;
    }

    auto prefetch = [&](int buf) {
        #pragma unroll
        for (int it = 0; it < 4; ++it) async_b128(aLds[it] + buf * BUFB, gA[it]);
        #pragma unroll
        for (int it = 0; it < 2; ++it) async_b128(bLds[it] + buf * (BUFB / 2), gB[it]);
        #pragma unroll
        for (int it = 0; it < 4; ++it) gA[it] += KP;
        #pragma unroll
        for (int it = 0; it < 2; ++it) gB[it] += KP;
    };

    v8f acc[2][2];
    #pragma unroll
    for (int mi = 0; mi < 2; ++mi)
        #pragma unroll
        for (int ni = 0; ni < 2; ++ni) acc[mi][ni] = vzero8();

    const int NS = Cdim / BK;
    prefetch(0);
    wait_async0();
    __syncthreads();

    for (int s = 0; s < NS; ++s) {
        const int p = s & 1;
        if (s + 1 < NS) prefetch(1 - p);

        FragBF aH[2], aL[2];
        #pragma unroll
        for (int mi = 0; mi < 2; ++mi) {
            int row = wm * 32 + mi * 16 + r;
            aH[mi].q[0] = *(const uint4*)&As[p][0][row][half * 4];
            aH[mi].q[1] = *(const uint4*)&As[p][0][row][half * 4 + 8];
            aL[mi].q[0] = *(const uint4*)&As[p][1][row][half * 4];
            aL[mi].q[1] = *(const uint4*)&As[p][1][row][half * 4 + 8];
        }
        #pragma unroll
        for (int ni = 0; ni < 2; ++ni) {
            FragBF bH, bL;
            int col = wn * 32 + ni * 16 + r;
            bH.q[0] = *(const uint4*)&Bs[p][0][col][half * 4];
            bH.q[1] = *(const uint4*)&Bs[p][0][col][half * 4 + 8];
            bL.q[0] = *(const uint4*)&Bs[p][1][col][half * 4];
            bL.q[1] = *(const uint4*)&Bs[p][1][col][half * 4 + 8];
            #pragma unroll
            for (int mi = 0; mi < 2; ++mi) {
                v8f c = acc[mi][ni];
                c = __builtin_amdgcn_wmma_f32_16x16x32_bf16(false, aH[mi].v, false, bH.v,
                                                            (short)0, c, false, false);
                c = __builtin_amdgcn_wmma_f32_16x16x32_bf16(false, aH[mi].v, false, bL.v,
                                                            (short)0, c, false, false);
                c = __builtin_amdgcn_wmma_f32_16x16x32_bf16(false, aL[mi].v, false, bH.v,
                                                            (short)0, c, false, false);
                acc[mi][ni] = c;
            }
        }
        wait_async0();
        __syncthreads();
    }

    #pragma unroll
    for (int mi = 0; mi < 2; ++mi) {
        #pragma unroll
        for (int ni = 0; ni < 2; ++ni) {
            int col = n0 + wn * 32 + ni * 16 + r;
            float bb = b2[col];
            #pragma unroll
            for (int v = 0; v < 8; ++v) {
                int row = m0 + wm * 32 + mi * 16 + half * 8 + v;
                out[(size_t)row * Cdim + col] = acc[mi][ni][v] + bb;
            }
        }
    }
}

// ---------------------------------------------------------------------------
extern "C" void kernel_launch(void* const* d_in, const int* in_sizes, int n_in,
                              void* d_out, int out_size, void* d_ws, size_t ws_size,
                              hipStream_t stream) {
    const float* query = (const float*)d_in[0];
    const int*   mask  = (const int*)d_in[3];
    const float* W1    = (const float*)d_in[4];
    const float* b1    = (const float*)d_in[5];
    const float* W2    = (const float*)d_in[6];
    const float* b2    = (const float*)d_in[7];
    const float* convw = (const float*)d_in[8];
    const float* bias  = (const float*)d_in[9];

    const int Bb = 16, T = 1024, C = 1024;
    const int M = Bb * T;
    const size_t MC  = (size_t)M * C;       // 16M elems
    const size_t MCp = MC / 2;              // pairs

    // workspace layout (4-byte units)
    float*    xg   = (float*)d_ws;                       // 64 MB fp32 GLU output
    unsigned* qhi  = (unsigned*)(xg + MC);               // 32 MB (aliased as yhi after conv)
    unsigned* qlo  = qhi + MCp;                          // 32 MB (aliased as ylo)
    unsigned* w1hi = qlo + MCp;                          // 4 MB
    unsigned* w1lo = w1hi + (size_t)2 * C * C / 2;       // 4 MB
    unsigned* w2hi = w1lo + (size_t)2 * C * C / 2;       // 2 MB
    unsigned* w2lo = w2hi + (size_t)C * C / 2;           // 2 MB
    float*    wsoft = (float*)(w2lo + (size_t)C * C / 2);

    lc_split<<<(int)(MCp / 256), 256, 0, stream>>>(query, qhi, qlo, (int)MCp);
    lc_split<<<(int)((size_t)C * C / 256), 256, 0, stream>>>(W1, w1hi, w1lo, C * C);
    lc_split<<<(int)((size_t)C * C / 2 / 256), 256, 0, stream>>>(W2, w2hi, w2lo, C * C / 2);
    lc_softmax_rows<<<1, 32, 0, stream>>>(convw, wsoft, 4, 31);

    lc_gemm1_glu<<<dim3(C / BN, M / BM), BDIM, 0, stream>>>(qhi, qlo, w1hi, w1lo, b1, xg, C);

    // q splits dead now; reuse their space for split conv output y
    unsigned* yhi = qhi;
    unsigned* ylo = qlo;
    lc_dwconv<<<dim3(T / CT, C / CC, Bb), 256, 0, stream>>>(xg, wsoft, bias, mask, yhi, ylo, T, C);

    lc_gemm2<<<dim3(C / BN, M / BM), BDIM, 0, stream>>>(yhi, ylo, w2hi, w2lo, b2, (float*)d_out, C);
}
